// VidCnn_35098472743353
// MI455X (gfx1250) — matrix operand order repaired and checked by reading, not compiled, not run on hardware
//
#include <hip/hip_runtime.h>

typedef _Float16 f16;
typedef __attribute__((ext_vector_type(16))) _Float16 v16h;
typedef __attribute__((ext_vector_type(8)))  float    v8f;
typedef __attribute__((ext_vector_type(4)))  unsigned int v4u;
typedef __attribute__((ext_vector_type(8)))  int      v8i;
typedef __attribute__((ext_vector_type(4)))  int      v4i;

// ---------------- geometry ----------------
constexpr int HW   = 160;        // input frame side
constexpr int PW   = 72;         // nn patch grid side
constexpr int PT   = 24;         // nn patch tile side (24x24 = 576 threads)
constexpr int DWD  = 38;         // diff tile side (24 + 14)
constexpr int SWD  = 112;        // staged strip width (38 + 74)
constexpr int SWP  = 114;        // padded strip width
constexpr int F146 = 146;
constexpr int F2   = 146 * 146;  // 21316
constexpr int VH   = 66;         // conv spatial side
constexpr int SP   = VH * VH;    // 4356
constexpr int KP   = 160;        // padded K for WMMA (150 -> 160 = 5*32)
constexpr int NT   = 273;        // ceil(4356/16)
constexpr int NPAD = NT * 16;    // 4368

// =====================================================================
// 1) NN search: one block per (ts, 24x24 patch tile). Box-filtered SSD,
//    per-thread sorted top-14 kept in registers.
// =====================================================================
__global__ __launch_bounds__(576) void nn_search_kernel(
    const float* __restrict__ seq, float* __restrict__ nnD, int* __restrict__ nnI)
{
    const int ts   = blockIdx.x;            // 0..6
    const int tile = blockIdx.y;            // 0..8
    const int bi0  = (tile / 3) * PT;
    const int bj0  = (tile % 3) * PT;
    const int tid  = threadIdx.x;
    const int li   = tid / PT, lj = tid % PT;

    __shared__ float sN[3][DWD][DWD];       // reference tile (frame 3)
    __shared__ f16   sS[3][DWD][SWP];       // staged search strip
    __shared__ float sD[DWD][DWD];          // per-displacement diff
    __shared__ float sR[DWD][PT];           // horizontal 15-sums

    for (int t = tid; t < 3 * DWD * DWD; t += 576) {
        int c = t / (DWD * DWD); int r = (t / DWD) % DWD; int q = t % DWD;
        sN[c][r][q] = seq[((c * 7 + 3) * HW + (37 + bi0 + r)) * HW + (37 + bj0 + q)];
    }
    __syncthreads();

    float bd[14]; int bix[14];
#pragma unroll
    for (int u = 0; u < 14; ++u) { bd[u] = 3.0e38f; bix[u] = 0; }

    for (int vs = 0; vs < 75; ++vs) {
        __syncthreads();
        for (int t = tid; t < 3 * DWD * SWD; t += 576) {
            int c = t / (DWD * SWD); int r = (t / SWD) % DWD; int q = t % SWD;
            sS[c][r][q] = (f16)seq[((c * 7 + ts) * HW + (vs + bi0 + r)) * HW + (bj0 + q)];
        }
        __syncthreads();
        for (int hv = 0; hv < 75; ++hv) {
            for (int t = tid; t < DWD * DWD; t += 576) {
                int r = t / DWD, q = t % DWD;
                float a0 = (float)sS[0][r][q + hv] - sN[0][r][q];
                float a1 = (float)sS[1][r][q + hv] - sN[1][r][q];
                float a2 = (float)sS[2][r][q + hv] - sN[2][r][q];
                sD[r][q] = (a0 * a0 + a1 * a1 + a2 * a2) * (1.0f / 3.0f);
            }
            __syncthreads();
            for (int t = tid; t < DWD * PT; t += 576) {
                int r = t / PT, j = t % PT;
                float s = 0.f;
#pragma unroll
                for (int dj = 0; dj < 15; ++dj) s += sD[r][j + dj];
                sR[r][j] = s;
            }
            __syncthreads();
            {
                float s = 0.f;
#pragma unroll
                for (int di = 0; di < 15; ++di) s += sR[li + di][lj];
                bool self = (ts == 3) && (vs == 37) && (hv == 37);
                if (!self && s < bd[13]) {
                    int gi = bi0 + li, gj = bj0 + lj;
                    int idx = ts * F2 + (gi + vs) * F146 + (gj + hv);
                    int u = 13;
                    while (u > 0 && bd[u - 1] > s) { bd[u] = bd[u - 1]; bix[u] = bix[u - 1]; --u; }
                    bd[u] = s; bix[u] = idx;
                }
            }
            __syncthreads();
        }
    }
    const int p = (bi0 + li) * PW + (bj0 + lj);
    for (int u = 0; u < 14; ++u) {
        nnD[(ts * (PW * PW) + p) * 14 + u] = bd[u];
        nnI[(ts * (PW * PW) + p) * 14 + u] = bix[u];
    }
}

// merge 7 sorted 14-lists per patch -> global top-14 (ascending distance)
__global__ void nn_merge_kernel(const float* __restrict__ nnD, const int* __restrict__ nnI,
                                int* __restrict__ mini)
{
    int p = blockIdx.x * blockDim.x + threadIdx.x;
    if (p >= PW * PW) return;
    int head[7];
#pragma unroll
    for (int t = 0; t < 7; ++t) head[t] = 0;
    for (int u = 0; u < 14; ++u) {
        float best = 3.2e38f; int bt = 0;   // > any stored distance (capped at 3.0e38f)
#pragma unroll
        for (int t = 0; t < 7; ++t) {
            if (head[t] < 14) {
                float v = nnD[(t * PW * PW + p) * 14 + head[t]];
                if (v < best) { best = v; bt = t; }
            }
        }
        mini[p * 14 + u] = nnI[(bt * PW * PW + p) * 14 + head[bt]];
        head[bt]++;
    }
}

// =====================================================================
// 2) create_layers: closed-form inverse of the mv/mh mosaic; one gather
//    per output element into x_f[n][147ch][66][66].
// =====================================================================
__global__ void build_layers_kernel(const float* __restrict__ seq, const int* __restrict__ mini,
                                    float* __restrict__ xf)
{
    long long gid = blockIdx.x * (long long)blockDim.x + threadIdx.x;
    const long long total = 15LL * 147 * SP;
    if (gid >= total) return;
    int X  = (int)(gid % VH);
    int Y  = (int)((gid / VH) % VH);
    int ch = (int)((gid / SP) % 147);
    int n  = (int)(gid / (147LL * SP));
    int k = ch / 3, c = ch % 3;
    int mv = k / 7, mh = k % 7;
    int t  = Y + 6 - mv; int iv = t / 7; int r = t % 7;
    int u  = X + 6 - mh; int ih = u / 7; int s = u % 7;
    int a  = mv + 7 * iv;   // 0..71
    int bb = mh + 7 * ih;   // 0..71
    int sel;
    if (n == 0) sel = 3 * F2 + (37 + a) * F146 + (37 + bb);
    else        sel = mini[(a * PW + bb) * 14 + (n - 1)];
    int fi  = sel / F2;  int rem = sel - fi * F2;
    int pv  = rem / F146; int ph = rem - pv * F146;
    // seq_c[c][fi][pv+r][ph+s] == seq_in[c][fi][pv+r+4][ph+s+4]
    float v = seq[((c * 7 + fi) * HW + (pv + r + 4)) * HW + (ph + s + 4)];
    xf[(n * 150 + ch) * SP + Y * VH + X] = v;
}

// wts channels 147..149: mean_k (layers[n,k] - layers[0,k])^2
__global__ void build_wts_kernel(float* __restrict__ xf)
{
    int gid = blockIdx.x * blockDim.x + threadIdx.x;
    if (gid >= 15 * 3 * SP) return;
    int p = gid % SP; int c = (gid / SP) % 3; int n = gid / (3 * SP);
    float acc = 0.f;
    for (int k = 0; k < 49; ++k) {
        float a = xf[(n * 150 + k * 3 + c) * SP + p];
        float b = xf[(0 * 150 + k * 3 + c) * SP + p];
        float d = a - b; acc += d * d;
    }
    xf[(n * 150 + 147 + c) * SP + p] = acc * (1.0f / 49.0f);
}

// =====================================================================
// 3) depthwise-ish 7x7 conv, reflect pad 3, groups of 3 channels (3->3)
// =====================================================================
__global__ void dw7x7_kernel(const float* __restrict__ in, const float* __restrict__ wvh,
                             float* __restrict__ out, int C)
{
    int gid = blockIdx.x * blockDim.x + threadIdx.x;
    if (gid >= C * SP) return;
    int p = gid % SP; int oc = gid / SP;
    int Y = p / VH, X = p % VH;
    int cb = (oc / 3) * 3;
    const float* w = wvh + oc * 147;
    float acc = 0.f;
    for (int i = 0; i < 3; ++i) {
        const float* ip = in + (cb + i) * SP;
#pragma unroll
        for (int ky = 0; ky < 7; ++ky) {
            int y = Y + ky - 3; y = (y < 0) ? -y : ((y > 65) ? 130 - y : y);
#pragma unroll
            for (int kx = 0; kx < 7; ++kx) {
                int x = X + kx - 3; x = (x < 0) ? -x : ((x > 65) ? 130 - x : x);
                acc += w[i * 49 + ky * 7 + kx] * ip[y * VH + x];
            }
        }
    }
    out[oc * SP + p] = acc;
}

// =====================================================================
// 4) f16 staging for the WMMA GEMM
//    Xh[g][p][k] (k contiguous -> B-frag = contiguous 32B), K,N zero-padded
//    Wh[g][m][k] row-major 160x160, zero-padded
// =====================================================================
__global__ void pack_x_kernel(const float* __restrict__ a, f16* __restrict__ xh, int G)
{
    int gid = blockIdx.x * blockDim.x + threadIdx.x;
    if (gid >= G * NPAD * KP) return;
    int k = gid % KP; int p = (gid / KP) % NPAD; int g = gid / (KP * NPAD);
    f16 v = (f16)0.f;
    if (p < SP && k < 150) v = (f16)a[(g * 150 + k) * SP + p];
    xh[((long long)g * NPAD + p) * KP + k] = v;
}

__global__ void pack_w_kernel(const float* __restrict__ wf, f16* __restrict__ wh, int G)
{
    int gid = blockIdx.x * blockDim.x + threadIdx.x;
    if (gid >= G * KP * KP) return;
    int k = gid % KP; int m = (gid / KP) % KP; int g = gid / (KP * KP);
    f16 v = (f16)0.f;
    if (m < 150 && k < 150) v = (f16)wf[(g * 150 + m) * 150 + k];
    wh[(g * KP + m) * KP + k] = v;
}

// =====================================================================
// 5) WMMA GEMM for w_f 1x1 group conv:  Y[g] = W[g](150x150) * X[g](150x4356)
//    block = (g, mtile) x 8 N-tile waves; A tile staged in LDS via TDM.
//    An opaque memory-clobber asm between the TDM issue and the fragment
//    reads makes the TDM's LDS write visible to the optimizer, so the
//    ds_load_b128 A reads are real but stay fast (non-volatile).
// =====================================================================
typedef union { v16h h; unsigned int u[8]; } frag16;

__global__ __launch_bounds__(256) void wf_gemm_kernel(
    const f16* __restrict__ wh, const f16* __restrict__ xh,
    float* __restrict__ y, int G)
{
    const int g     = blockIdx.x / 10;
    const int mtile = blockIdx.x % 10;
    const int wave  = threadIdx.x >> 5;
    const int lane  = threadIdx.x & 31;
    const int lhalf = lane >> 4;        // 0: lanes 0-15, 1: lanes 16-31
    const int lmod  = lane & 15;

    __shared__ f16 shA[16 * KP];        // 16 x 160 A tile (5 KB)

    const f16* Abase = wh + ((long long)g * KP + mtile * 16) * KP;

#if defined(__has_builtin) && __has_builtin(__builtin_amdgcn_tensor_load_to_lds)
    if (wave == 0) {
        unsigned long long ga = (unsigned long long)Abase;
        v4u d0; v8i d1; v4i d2 = {0, 0, 0, 0}; v4i d3 = {0, 0, 0, 0};
        v8i d4 = {0, 0, 0, 0, 0, 0, 0, 0};                   // 6-arg form (clang-23 lane)
        d0[0] = 1u;                                          // count=1, gather off
        d0[1] = 0u;                                          // lds_addr: shA is first LDS object
        d0[2] = (unsigned)(ga & 0xFFFFFFFFull);              // global addr lo
        d0[3] = (unsigned)((ga >> 32) & 0x1FFFFFFull) | (2u << 30); // addr hi | type=2
        d1[0] = (int)(1u << 16);                             // data_size = 2 bytes
        d1[1] = (int)(160u << 16);                           // tensor_dim0[15:0] = 160
        d1[2] = (int)(16u << 16);                            // dim0 hi=0 | tensor_dim1 = 16
        d1[3] = (int)(160u << 16);                           // dim1 hi=0 | tile_dim0 = 160
        d1[4] = 16;                                          // tile_dim1 = 16, tile_dim2 = 0
        d1[5] = 160;                                         // tensor_dim0_stride lo
        d1[6] = 0; d1[7] = 0;
        __builtin_amdgcn_tensor_load_to_lds(d0, d1, d2, d3, d4, 0);
        __builtin_amdgcn_s_wait_tensorcnt(0);
    }
    // Opaque writer: TDM deposited the tile in LDS behind the optimizer's
    // back; this clobber forbids undef-folding of the shA reads below.
    asm volatile("" : : "s"(&shA[0]) : "memory");
    __syncthreads();
#else
    for (int t = threadIdx.x; t < 16 * KP; t += 256) shA[t] = Abase[t];
    __syncthreads();
#endif

    int ntile = blockIdx.y * 8 + wave;
    if (ntile >= NT) return;
    const int p = ntile * 16 + lmod;                 // column (spatial), < 4368
    const f16* Bbase = xh + ((long long)g * NPAD + p) * KP;
    __builtin_prefetch(Bbase, 0, 1);                 // global_prefetch_b8

    v8f acc = {};
#pragma unroll
    for (int kc = 0; kc < 5; ++kc) {
        const int k0 = kc * 32;
        frag16 A, B;
        // A frag: lanes<16 hold K = k0..k0+15, lanes>=16 hold K = k0+16..k0+31 (contiguous)
        // two contiguous 16B chunks -> ds_load_b128 x2
        const uint4* a0 = (const uint4*)&shA[lmod * KP + k0 + lhalf * 16];
        const uint4* a1 = (const uint4*)&shA[lmod * KP + k0 + lhalf * 16 + 8];
        ((uint4*)A.u)[0] = *a0;
        ((uint4*)A.u)[1] = *a1;
        // B frag: lanes<16 hold K = k0..k0+15, lanes>=16 hold K = k0+16..k0+31 (contiguous)
        const uint4* b0 = (const uint4*)&Bbase[k0 + lhalf * 16];
        ((uint4*)B.u)[0] = b0[0];
        ((uint4*)B.u)[1] = b0[1];
        acc = __builtin_amdgcn_wmma_f32_16x16x32_f16(false, A.h, false, B.h,
                                                     (short)0, acc, false, false);
    }
    // D layout: VGPR r -> M = r (lanes<16) / 8+r (lanes>=16), N = lane&15
#pragma unroll
    for (int r = 0; r < 8; ++r) {
        int m  = mtile * 16 + r + lhalf * 8;
        int pc = ntile * 16 + lmod;
        if (m < 150 && pc < SP)
            y[((long long)g * 150 + m) * SP + pc] = acc[r];
    }
}

// =====================================================================
// 6) w_n mixer: out[j][fc] = sum_i Wn[fc*no+j][i] * Y[i][fc]  (+bias,relu)
// =====================================================================
__global__ void wn_kernel(const float* __restrict__ yf, const float* __restrict__ wn,
                          const float* __restrict__ bias, float* __restrict__ out,
                          int ni, int no, int dorelu)
{
    int gid = blockIdx.x * blockDim.x + threadIdx.x;
    if (gid >= 150 * SP) return;
    int p = gid % SP; int fc = gid / SP;
    for (int j = 0; j < no; ++j) {
        float acc = 0.f;
        for (int i = 0; i < ni; ++i)
            acc += wn[(fc * no + j) * ni + i] * yf[(i * 150 + fc) * SP + p];
        if (bias) acc += bias[j * 150 + fc];
        if (dorelu) acc = fmaxf(acc, 0.f);
        out[(j * 150 + fc) * SP + p] = acc;
    }
}

// =====================================================================
// 7) train-mode BN (+relu), one block per channel, two-pass in-block
// =====================================================================
__global__ __launch_bounds__(256) void bn_relu_kernel(float* __restrict__ x,
                                                      const float* __restrict__ gamma,
                                                      const float* __restrict__ beta)
{
    int ch = blockIdx.x;
    float* xp = x + (long long)ch * SP;
    __shared__ float ssum[8], ssq[8];
    float s = 0.f, q = 0.f;
    for (int p = threadIdx.x; p < SP; p += 256) { float v = xp[p]; s += v; q += v * v; }
    for (int off = 16; off > 0; off >>= 1) {
        s += __shfl_down(s, off, 32);
        q += __shfl_down(q, off, 32);
    }
    int wv = threadIdx.x >> 5, ln = threadIdx.x & 31;
    if (ln == 0) { ssum[wv] = s; ssq[wv] = q; }
    __syncthreads();
    if (threadIdx.x == 0) {
        float S = 0.f, Q = 0.f;
        for (int i = 0; i < 8; ++i) { S += ssum[i]; Q += ssq[i]; }
        float m = S / (float)SP;
        ssum[0] = m;
        ssq[0]  = Q / (float)SP - m * m;
    }
    __syncthreads();
    float m = ssum[0], var = ssq[0];
    float sc = gamma[ch] * rsqrtf(var + 1e-5f);
    float sh = beta[ch] - m * sc;
    for (int p = threadIdx.x; p < SP; p += 256)
        xp[p] = fmaxf(xp[p] * sc + sh, 0.f);
}

// =====================================================================
// 8) final: out = valid - (w_f4 * t + b4)
// =====================================================================
__global__ void final_kernel(const float* __restrict__ t, const float* __restrict__ wf4,
                             const float* __restrict__ b4, const float* __restrict__ seq,
                             float* __restrict__ out)
{
    int gid = blockIdx.x * blockDim.x + threadIdx.x;
    if (gid >= 3 * SP) return;
    int p = gid % SP; int c = gid / SP;
    int Y = p / VH, X = p % VH;
    float acc = b4[c];
    for (int fc = 0; fc < 150; ++fc) acc += wf4[c * 150 + fc] * t[fc * SP + p];
    float valid = seq[((c * 7 + 3) * HW + (47 + Y)) * HW + (47 + X)];
    out[c * SP + p] = valid - acc;
}

// =====================================================================
// host launcher
// =====================================================================
extern "C" void kernel_launch(void* const* d_in, const int* in_sizes, int n_in,
                              void* d_out, int out_size, void* d_ws, size_t ws_size,
                              hipStream_t stream)
{
    (void)in_sizes; (void)n_in; (void)out_size; (void)ws_size;
    const float* seq = (const float*)d_in[0];
    const float* wvh[5] = { (const float*)d_in[2],  (const float*)d_in[5],
                            (const float*)d_in[8],  (const float*)d_in[11],
                            (const float*)d_in[21] };
    const float* wfp[5] = { (const float*)d_in[3],  (const float*)d_in[6],
                            (const float*)d_in[9],  (const float*)d_in[12],
                            (const float*)d_in[22] };
    const float* wnp[4] = { (const float*)d_in[4],  (const float*)d_in[7],
                            (const float*)d_in[10], (const float*)d_in[13] };
    const float* b0  = (const float*)d_in[14];
    const float* gg[3] = { (const float*)d_in[15], (const float*)d_in[17], (const float*)d_in[19] };
    const float* bb[3] = { (const float*)d_in[16], (const float*)d_in[18], (const float*)d_in[20] };
    const float* b4  = (const float*)d_in[23];
    float* out = (float*)d_out;

    char* ws = (char*)d_ws;
    size_t off = 0;
    auto carve = [&](size_t bytes) -> char* {
        char* p = ws + off;
        off = (off + bytes + 255) & ~(size_t)255;
        return p;
    };
    float* nnD  = (float*)carve(7ull * PW * PW * 14 * 4);
    int*   nnI  = (int*)  carve(7ull * PW * PW * 14 * 4);
    int*   mini = (int*)  carve((size_t)PW * PW * 14 * 4);
    float* X    = (float*)carve(15ull * 150 * SP * 4);   // activations (in-place across layers)
    float* T    = (float*)carve(15ull * 150 * SP * 4);   // depthwise temp
    float* Yf   = (float*)carve(15ull * 150 * SP * 4);   // GEMM output
    f16*   Xh   = (f16*)  carve(15ull * NPAD * KP * 2);  // f16 activations [g][p][k]
    f16*   Wh   = (f16*)  carve(15ull * KP * KP * 2);    // f16 weights [g][m][k]

    // --- NN search + merge ---
    nn_search_kernel<<<dim3(7, 9), 576, 0, stream>>>(seq, nnD, nnI);
    nn_merge_kernel<<<(PW * PW + 255) / 256, 256, 0, stream>>>(nnD, nnI, mini);

    // --- layers + wts -> x_f ---
    {
        long long tot = 15LL * 147 * SP;
        build_layers_kernel<<<(unsigned)((tot + 255) / 256), 256, 0, stream>>>(seq, mini, X);
    }
    build_wts_kernel<<<(15 * 3 * SP + 255) / 256, 256, 0, stream>>>(X);

    // --- sep_conv_net: 4 sep_fm layers ---
    const int nis[4] = {15, 8, 4, 2};
    const int nos[4] = {8, 4, 2, 1};
    for (int l = 0; l < 4; ++l) {
        int ni = nis[l], no = nos[l];
        int C = ni * 150;
        dw7x7_kernel<<<(C * SP + 255) / 256, 256, 0, stream>>>(X, wvh[l], T, C);
        pack_x_kernel<<<(ni * NPAD * KP + 255) / 256, 256, 0, stream>>>(T, Xh, ni);
        pack_w_kernel<<<(ni * KP * KP + 255) / 256, 256, 0, stream>>>(wfp[l], Wh, ni);
        wf_gemm_kernel<<<dim3(ni * 10, (NT + 7) / 8), 256, 0, stream>>>(Wh, Xh, Yf, ni);
        wn_kernel<<<(150 * SP + 255) / 256, 256, 0, stream>>>(
            Yf, wnp[l], (l == 0) ? b0 : nullptr, X, ni, no, (l == 0) ? 1 : 0);
        if (l > 0)
            bn_relu_kernel<<<no * 150, 256, 0, stream>>>(X, gg[l - 1], bb[l - 1]);
    }

    // --- final layer ---
    dw7x7_kernel<<<(150 * SP + 255) / 256, 256, 0, stream>>>(X, wvh[4], T, 150);
    final_kernel<<<(3 * SP + 255) / 256, 256, 0, stream>>>(T, wfp[4], b4, seq, out);
}
